// TopoLoss_16406775071323
// MI455X (gfx1250) — compile-verified, matrix operand order checked by reading, not run on hardware
//
#include <hip/hip_runtime.h>
#include <hip/hip_bf16.h>
#include <math.h>

// Problem constants from reference setup_inputs(): B=64, N=1024, D=3.
#define TB   64
#define TN   1024
#define TD   3
#define NTHR 256

typedef __attribute__((ext_vector_type(16))) _Float16 v16h;
typedef __attribute__((ext_vector_type(8)))  float    v8f;

// ---------------------------------------------------------------------------
// Kernel 0: zero the scalar output (harness poisons d_out with 0xAA).
// ---------------------------------------------------------------------------
__global__ void topo_zero_out(float* out) {
    if (blockIdx.x == 0 && threadIdx.x == 0) out[0] = 0.0f;
}

// ---------------------------------------------------------------------------
// Kernel 1: WMMA pairwise-distance builder.
// One wave computes one 16x16 tile of D for one cloud via
// v_wmma_f32_16x16x32_f16 (Gram trick, K=3 padded to 32 with zeros).
// d[i][j] = sqrt(max(|xi|^2 + |xj|^2 - 2*<xi,xj>, 0) + 1e-12), norms in f32.
// 128 clouds (gts then preds) * 64x64 tiles = 524288 waves; 8 waves/block.
// ---------------------------------------------------------------------------
__global__ __launch_bounds__(NTHR)
void topo_wmma_dist(const float* __restrict__ gts,
                    const float* __restrict__ preds,
                    float* __restrict__ Dg) {
    const int gwave = (blockIdx.x * NTHR + threadIdx.x) >> 5;
    const int lane  = threadIdx.x & 31;
    const int cloud = gwave >> 12;          // 4096 tiles per cloud
    const int t     = gwave & 4095;
    const int ti    = t >> 6;
    const int tj    = t & 63;
    const int half  = lane >> 4;
    const int l15   = lane & 15;

    const float* P = (cloud < TB ? gts : preds) + (size_t)(cloud & (TB - 1)) * TN * TD;

    // A fragment: 16x32 f16, lane(0..15)=row M, v16h elems 0..7 = K 0..7
    // (half 0) / K 8..15 (half 1). Only K=0..2 are nonzero -> half 1 all 0.
    // B fragment: 32x16 f16, lane(0..15)=col N, elems 0..7 = K 0..7, 8..15 = K 16..23
    // (half 0) / K 8..15, 24..31 (half 1, all zero here).
    v16h a = {};
    v16h bm = {};
    if (half == 0) {
        const float* rp = P + (size_t)(ti * 16 + l15) * TD;
        a[0] = (_Float16)rp[0]; a[1] = (_Float16)rp[1]; a[2] = (_Float16)rp[2];
        const float* cp = P + (size_t)(tj * 16 + l15) * TD;
        bm[0] = (_Float16)cp[0]; bm[1] = (_Float16)cp[1]; bm[2] = (_Float16)cp[2];
    }
    v8f c = {};
    c = __builtin_amdgcn_wmma_f32_16x16x32_f16(
        /*neg_a=*/false, a, /*neg_b=*/false, bm,
        /*c_mod=*/(short)0, c, /*reuse_a=*/false, /*reuse_b=*/false);

    // f32 norm corrections + sqrt, then store the tile.
    const int colg = tj * 16 + l15;
    const float* cp2 = P + (size_t)colg * TD;
    const float cx = cp2[0], cy = cp2[1], cz = cp2[2];
    const float sqc = cx * cx + cy * cy + cz * cz;
    float* Dout = Dg + (size_t)cloud * TN * TN;
#pragma unroll
    for (int r = 0; r < 8; ++r) {
        const int rowg = ti * 16 + half * 8 + r;   // C/D layout: half selects M+8
        const float* rp = P + (size_t)rowg * TD;
        const float rx = rp[0], ry = rp[1], rz = rp[2];
        const float sqr = rx * rx + ry * ry + rz * rz;
        const float d2  = sqr + sqc - 2.0f * c[r];
        Dout[(size_t)rowg * TN + colg] = sqrtf(fmaxf(d2, 0.0f) + 1e-12f);
    }
}

// ---------------------------------------------------------------------------
// Kernel 2: per-batch Prim MST -> sorted deaths (both clouds) -> W1 -> atomic.
// One block per batch element; everything LDS-resident (~27 KB / 320 KB WGP).
// useD=1: read rows of the WMMA-built D; useD=0: 3-FMA+sqrt on the fly.
// ---------------------------------------------------------------------------
__global__ __launch_bounds__(NTHR)
void topo_prim_loss(const float* __restrict__ gts,
                    const float* __restrict__ preds,
                    const float* __restrict__ Dg,
                    float* __restrict__ out,
                    int useD) {
    __shared__ float px[TN], py[TN], pz[TN];
    __shared__ float mind[TN];
    __shared__ unsigned char intree[TN];
    __shared__ float dth[2][TN];
    __shared__ float rval[NTHR];
    __shared__ int   ridx[NTHR];

    const int tid = threadIdx.x;
    const int b   = blockIdx.x;
    const float INFV = __builtin_inff();

    for (int c = 0; c < 2; ++c) {
        const float* P    = (c == 0 ? gts : preds) + (size_t)b * TN * TD;
        const float* Dmat = Dg + (size_t)(c * TB + b) * TN * TN;

        for (int i = tid; i < TN; i += NTHR) {
            px[i] = P[i * 3 + 0];
            py[i] = P[i * 3 + 1];
            pz[i] = P[i * 3 + 2];
            intree[i] = 0;
        }
        __syncthreads();

        // min_dist0 = D[0][:]  (node 0 in tree)
        const float x0 = px[0], y0 = py[0], z0 = pz[0];
        for (int i = tid; i < TN; i += NTHR) {
            float d;
            if (useD) {
                d = Dmat[i];
            } else {
                const float dx = px[i] - x0, dy = py[i] - y0, dz = pz[i] - z0;
                d = sqrtf(fmaxf(dx * dx + dy * dy + dz * dz, 0.0f) + 1e-12f);
            }
            mind[i] = d;
        }
        if (tid == 0) intree[0] = 1;
        __syncthreads();

        for (int it = 0; it < TN - 1; ++it) {
            // masked argmin (first-index tie-break, matching jnp.argmin)
            float lv = INFV;
            int   li = 0;
            for (int i = tid; i < TN; i += NTHR) {
                const float v = intree[i] ? INFV : mind[i];
                if (v < lv) { lv = v; li = i; }
            }
            rval[tid] = lv;
            ridx[tid] = li;
            __syncthreads();
            for (int s = NTHR / 2; s > 0; s >>= 1) {
                if (tid < s) {
                    const float v2 = rval[tid + s];
                    const int   i2 = ridx[tid + s];
                    if (v2 < rval[tid] || (v2 == rval[tid] && i2 < ridx[tid])) {
                        rval[tid] = v2; ridx[tid] = i2;
                    }
                }
                __syncthreads();
            }
            const int   j     = ridx[0];
            const float death = rval[0];
            const float jx = px[j], jy = py[j], jz = pz[j];
            if (tid == 0) { intree[j] = 1; dth[c][it] = death; }
            // relax: min_dist = min(min_dist, D[j][:]) (unmasked, as in ref)
            const float* Dj = Dmat + (size_t)j * TN;
            for (int i = tid; i < TN; i += NTHR) {
                float d;
                if (useD) {
                    d = Dj[i];
                } else {
                    const float dx = px[i] - jx, dy = py[i] - jy, dz = pz[i] - jz;
                    d = sqrtf(fmaxf(dx * dx + dy * dy + dz * dz, 0.0f) + 1e-12f);
                }
                mind[i] = fminf(mind[i], d);
            }
            __syncthreads();
        }

        // bitonic sort of 1023 deaths (pad slot 1023 with +inf)
        if (tid == 0) dth[c][TN - 1] = INFV;
        __syncthreads();
        float* arr = dth[c];
        for (int k = 2; k <= TN; k <<= 1) {
            for (int jj = k >> 1; jj > 0; jj >>= 1) {
                for (int idx = tid; idx < TN; idx += NTHR) {
                    const int l = idx ^ jj;
                    if (l > idx) {
                        const float a0 = arr[idx], a1 = arr[l];
                        const bool asc = ((idx & k) == 0);
                        if ((a0 > a1) == asc) { arr[idx] = a1; arr[l] = a0; }
                    }
                }
                __syncthreads();
            }
        }
        __syncthreads();
    }

    // W1 of the two sorted diagrams (births all 0, equal cardinality)
    float s = 0.0f;
    for (int i = tid; i < TN - 1; i += NTHR) s += fabsf(dth[0][i] - dth[1][i]);
    rval[tid] = s;
    __syncthreads();
    for (int st = NTHR / 2; st > 0; st >>= 1) {
        if (tid < st) rval[tid] += rval[tid + st];
        __syncthreads();
    }
    if (tid == 0) atomicAdd(out, rval[0] * (1.0f / (float)TB));  // mean * lambda(=1)
}

// ---------------------------------------------------------------------------
extern "C" void kernel_launch(void* const* d_in, const int* in_sizes, int n_in,
                              void* d_out, int out_size, void* d_ws, size_t ws_size,
                              hipStream_t stream) {
    (void)in_sizes; (void)n_in; (void)out_size;
    const float* gts   = (const float*)d_in[0];
    const float* preds = (const float*)d_in[1];
    float* out = (float*)d_out;
    float* Dg  = (float*)d_ws;

    // Workspace needed to materialize both distance stacks: 2*64*1024*1024 f32.
    const size_t needD = (size_t)2 * TB * (size_t)TN * (size_t)TN * sizeof(float);
    const int useD = (ws_size >= needD) ? 1 : 0;

    topo_zero_out<<<1, 32, 0, stream>>>(out);

    if (useD) {
        // 2*64 clouds * 4096 tiles, 8 waves (tiles) per 256-thread block
        const int total_waves = 2 * TB * 64 * 64;
        const int blocks = total_waves / (NTHR / 32);
        topo_wmma_dist<<<blocks, NTHR, 0, stream>>>(gts, preds, Dg);
    }

    topo_prim_loss<<<TB, NTHR, 0, stream>>>(gts, preds, Dg, out, useD);
}
